// GINLayer_89635967467763
// MI455X (gfx1250) — matrix operand order repaired and checked by reading, not compiled, not run on hardware
//
#include <hip/hip_runtime.h>

typedef __attribute__((ext_vector_type(16))) __bf16 v16bf;
typedef __attribute__((ext_vector_type(8)))  float  v8f;

#define N_NODES 10000
#define N_EDGES 640000
#define D 128

// ---------------------------------------------------------------------------
// Kernel 1: agg = x   (self term of GIN, and resets the atomic accumulator
// every launch so graph replays are deterministic)
// ---------------------------------------------------------------------------
__global__ __launch_bounds__(256) void gin_init_agg(const float4* __restrict__ x,
                                                    float4* __restrict__ agg) {
    int i = blockIdx.x * blockDim.x + threadIdx.x;   // N_NODES*D/4 threads
    agg[i] = x[i];
}

// ---------------------------------------------------------------------------
// Kernel 2: agg[dst] += x[src] for every edge.
// One wave (32 lanes) per edge: each lane loads a float4 of the 512B source
// row (coalesced, L2-resident) and scatters 4 f32 atomics.
// ---------------------------------------------------------------------------
__global__ __launch_bounds__(256) void gin_scatter(const float* __restrict__ x,
                                                   const int* __restrict__ src,
                                                   const int* __restrict__ dst,
                                                   float* __restrict__ agg) {
    int e    = blockIdx.x * 8 + (threadIdx.x >> 5);
    int lane = threadIdx.x & 31;
    int s = src[e];
    int d = dst[e];
    float4 v = ((const float4*)(x + (size_t)s * D))[lane];
    float* o = agg + (size_t)d * D + lane * 4;
    atomicAdd(o + 0, v.x);
    atomicAdd(o + 1, v.y);
    atomicAdd(o + 2, v.z);
    atomicAdd(o + 3, v.w);
}

// ---------------------------------------------------------------------------
// Kernel 3: fused MLP  out = relu(agg@W1 + b1) @ W2 + b2
// Block = 8 waves = one 16-row node slab. Wave w owns output columns
// [16w, 16w+16). Stage-1 result parked in LDS as bf16, consumed by stage 2.
// ---------------------------------------------------------------------------
__global__ __launch_bounds__(256) void gin_mlp(const float* __restrict__ agg,
                                               const float* __restrict__ W1,
                                               const float* __restrict__ b1,
                                               const float* __restrict__ W2,
                                               const float* __restrict__ b2,
                                               float* __restrict__ out) {
    __shared__ __bf16 hs[16][D + 8];   // +8 bf16 pad to stagger banks

    const int wave = threadIdx.x >> 5;     // 0..7 -> column tile
    const int lane = threadIdx.x & 31;
    const int half = lane >> 4;            // 0 or 1
    const int mr   = lane & 15;            // A-matrix row within tile
    const int nc   = lane & 15;            // B/C/D column within tile
    const int m0   = blockIdx.x * 16;      // node row base
    const int n0   = wave * 16;            // output column base

    // ---- stage 1: h = relu((x + sum_msgs) @ W1 + b1) ----
    v8f acc = {};
    for (int k0 = 0; k0 < D; k0 += 32) {
        v16bf a, b;
        // A: 16-bit 16x32 layout. lane half selects K base offset of 8.
        const float* arow = agg + (size_t)(m0 + mr) * D + k0 + 8 * half;
#pragma unroll
        for (int i = 0; i < 4; ++i) {
            a[2 * i]     = (__bf16)arow[2 * i];
            a[2 * i + 1] = (__bf16)arow[2 * i + 1];
        }
#pragma unroll
        for (int i = 4; i < 8; ++i) {
            a[2 * i]     = (__bf16)arow[16 + 2 * (i - 4)];
            a[2 * i + 1] = (__bf16)arow[16 + 2 * (i - 4) + 1];
        }
        // B: K-pairs per VGPR; same K order as A (lane half offsets K by 8, elems 8..15 at +16)
        const float* wcol = W1 + (size_t)(k0 + 8 * half) * D + n0 + nc;
#pragma unroll
        for (int i = 0; i < 8; ++i) {
            b[2 * i]     = (__bf16)wcol[(2 * i + (i >= 4 ? 8 : 0)) * D];
            b[2 * i + 1] = (__bf16)wcol[(2 * i + 1 + (i >= 4 ? 8 : 0)) * D];
        }
        acc = __builtin_amdgcn_wmma_f32_16x16x32_bf16(false, a, false, b,
                                                      (short)0, acc, false, false);
    }

    // bias + relu, stash slab in LDS as bf16 (C/D layout: row r+8*half, col nc)
    float bias1 = b1[n0 + nc];
#pragma unroll
    for (int r = 0; r < 8; ++r) {
        float h = acc[r] + bias1;
        h = h > 0.0f ? h : 0.0f;
        hs[r + 8 * half][n0 + nc] = (__bf16)h;
    }
    __syncthreads();

    // ---- stage 2: out = h @ W2 + b2 ----
    v8f acc2 = {};
    for (int k0 = 0; k0 < D; k0 += 32) {
        v16bf a, b;
        const __bf16* arow = &hs[mr][k0 + 8 * half];
#pragma unroll
        for (int i = 0; i < 4; ++i) {
            a[2 * i]     = arow[2 * i];
            a[2 * i + 1] = arow[2 * i + 1];
        }
#pragma unroll
        for (int i = 4; i < 8; ++i) {
            a[2 * i]     = arow[16 + 2 * (i - 4)];
            a[2 * i + 1] = arow[16 + 2 * (i - 4) + 1];
        }
        const float* wcol = W2 + (size_t)(k0 + 8 * half) * D + n0 + nc;
#pragma unroll
        for (int i = 0; i < 8; ++i) {
            b[2 * i]     = (__bf16)wcol[(2 * i + (i >= 4 ? 8 : 0)) * D];
            b[2 * i + 1] = (__bf16)wcol[(2 * i + 1 + (i >= 4 ? 8 : 0)) * D];
        }
        acc2 = __builtin_amdgcn_wmma_f32_16x16x32_bf16(false, a, false, b,
                                                       (short)0, acc2, false, false);
    }

    float bias2 = b2[n0 + nc];
#pragma unroll
    for (int r = 0; r < 8; ++r) {
        out[(size_t)(m0 + r + 8 * half) * D + n0 + nc] = acc2[r] + bias2;
    }
}

// ---------------------------------------------------------------------------
extern "C" void kernel_launch(void* const* d_in, const int* in_sizes, int n_in,
                              void* d_out, int out_size, void* d_ws, size_t ws_size,
                              hipStream_t stream) {
    const float* x  = (const float*)d_in[0];
    const int*   ei = (const int*)d_in[1];     // [2, E]: src row then dst row
    const float* W1 = (const float*)d_in[2];
    const float* b1 = (const float*)d_in[3];
    const float* W2 = (const float*)d_in[4];
    const float* b2 = (const float*)d_in[5];
    float* out = (float*)d_out;
    float* agg = (float*)d_ws;                 // N_NODES*D f32 = 5.12 MB

    gin_init_agg<<<(N_NODES * D / 4) / 256, 256, 0, stream>>>((const float4*)x,
                                                              (float4*)agg);
    gin_scatter<<<N_EDGES / 8, 256, 0, stream>>>(x, ei, ei + N_EDGES, agg);
    gin_mlp<<<N_NODES / 16, 256, 0, stream>>>(agg, W1, b1, W2, b2, out);
}
